// Non_local_Block_51539608036
// MI455X (gfx1250) — matrix operand order repaired and checked by reading, not compiled, not run on hardware
//
#include <hip/hip_runtime.h>

// Non-local block (B=8, C=192, O=96, H=W=64, N=4096) for gfx1250.
// All GEMMs via v_wmma_f32_16x16x32_bf16; attention fused (flash-style,
// never materializes the 4096x4096 score matrix).

#define C_DIM 192
#define O_DIM 96
#define N_DIM 4096
#define B_DIM 8

typedef __attribute__((ext_vector_type(16))) __bf16 v16bf;
typedef __attribute__((ext_vector_type(8)))  float  v8f;

union BF16x16 { v16bf v; uint4 q[2]; };

// A-fragment (16x32 bf16) from row-major [M][K] matrix, leading dim ld.
// ISA layout: lane l holds row (l&15); k runs [kb, kb+8) and [kb+16, kb+24),
// kb = k0 + (l>>4)*8. Two 16-byte contiguous loads per lane.
__device__ __forceinline__ v16bf load_a_frag(const __bf16* p, int ld, int row0, int k0) {
  int lane = threadIdx.x & 31;
  const __bf16* q = p + (size_t)(row0 + (lane & 15)) * ld + k0 + ((lane >> 4) << 3);
  BF16x16 t;
  t.q[0] = *reinterpret_cast<const uint4*>(q);
  t.q[1] = *reinterpret_cast<const uint4*>(q + 16);
  return t.v;
}

// B-fragment (32x16 bf16) where B[k][col] = p[col*ld + k] (i.e. B^T stored
// row-major). ISA layout: lane l holds col (l&15); k runs [kb, kb+8) and
// [kb+16, kb+24), kb = k0 + (l>>4)*8. Two 16-byte contiguous loads per lane.
__device__ __forceinline__ v16bf load_bt_frag(const __bf16* p, int ld, int col0, int k0) {
  int lane = threadIdx.x & 31;
  const __bf16* q = p + (size_t)(col0 + (lane & 15)) * ld + k0 + ((lane >> 4) << 3);
  BF16x16 t;
  t.q[0] = *reinterpret_cast<const uint4*>(q);
  t.q[1] = *reinterpret_cast<const uint4*>(q + 16);
  return t.v;
}

__device__ __forceinline__ v8f wmma_bf16(v16bf a, v16bf b, v8f c) {
  return __builtin_amdgcn_wmma_f32_16x16x32_bf16(false, a, false, b, (short)0, c,
                                                 false, false);
}

// ---------------- fp32 -> bf16 weight conversion ----------------
__global__ void cvt_f32_bf16(const float* __restrict__ s, __bf16* __restrict__ d, int n) {
  int i = blockIdx.x * blockDim.x + threadIdx.x;
  if (i < n) d[i] = (__bf16)s[i];
}

// ---------------- x [B][C][N] f32 -> xT [B][N][C] bf16 ----------------
__global__ void transpose_x(const float* __restrict__ x, __bf16* __restrict__ xT) {
  __shared__ __bf16 tile[32][33];
  int b = blockIdx.z;
  int c0 = blockIdx.y * 32;
  int n0 = blockIdx.x * 32;
  int tx = threadIdx.x;   // 0..31
  int ty = threadIdx.y;   // 0..7
  const float* xb = x + ((size_t)b * C_DIM + c0) * N_DIM + n0;
#pragma unroll
  for (int i = 0; i < 4; ++i) {
    int c = ty + i * 8;
    tile[c][tx] = (__bf16)xb[(size_t)c * N_DIM + tx];
  }
  __syncthreads();
  __bf16* xtb = xT + ((size_t)b * N_DIM + n0) * C_DIM + c0;
#pragma unroll
  for (int i = 0; i < 4; ++i) {
    int n = ty + i * 8;
    xtb[(size_t)n * C_DIM + tx] = tile[tx][n];
  }
}

// ---------------- projections: theta/phi [B][N][O], g [B][O][N] ----------------
__global__ void __launch_bounds__(128)
proj_kernel(const __bf16* __restrict__ xT,
            const __bf16* __restrict__ wTheta, const float* __restrict__ bTheta,
            const __bf16* __restrict__ wPhi,   const float* __restrict__ bPhi,
            const __bf16* __restrict__ wG,     const float* __restrict__ bG,
            __bf16* __restrict__ theta, __bf16* __restrict__ phi,
            __bf16* __restrict__ g) {
  int b = blockIdx.y;
  int wave = threadIdx.x >> 5, lane = threadIdx.x & 31;
  int half = lane >> 4, col = lane & 15;
  int n0 = blockIdx.x * 64 + wave * 16;
  const __bf16* xb = xT + (size_t)b * N_DIM * C_DIM;
  const v8f vzero = {};

  // xT strip reused as A for theta/phi (out = xT * W^T -> [n][o])
  v16bf xa[6];
#pragma unroll
  for (int kk = 0; kk < 6; ++kk) xa[kk] = load_a_frag(xb, C_DIM, n0, kk * 32);

  for (int sel = 0; sel < 2; ++sel) {
    const __bf16* w    = sel ? wPhi : wTheta;
    const float*  bias = sel ? bPhi : bTheta;
    __bf16* dst = (sel ? phi : theta) + (size_t)b * N_DIM * O_DIM;
#pragma unroll
    for (int ot = 0; ot < 6; ++ot) {
      v8f acc = vzero;
#pragma unroll
      for (int kk = 0; kk < 6; ++kk)
        acc = wmma_bf16(xa[kk], load_bt_frag(w, C_DIM, ot * 16, kk * 32), acc);
      float bv = bias[ot * 16 + col];
#pragma unroll
      for (int r = 0; r < 8; ++r) {
        int n = n0 + half * 8 + r;
        dst[(size_t)n * O_DIM + ot * 16 + col] = (__bf16)(acc[r] + bv);
      }
    }
  }
  // g = W * x -> [o][n]
  __bf16* gb = g + (size_t)b * O_DIM * N_DIM;
#pragma unroll
  for (int ot = 0; ot < 6; ++ot) {
    v8f acc = vzero;
#pragma unroll
    for (int kk = 0; kk < 6; ++kk)
      acc = wmma_bf16(load_a_frag(wG, C_DIM, ot * 16, kk * 32),
                      load_bt_frag(xb, C_DIM, n0, kk * 32), acc);
#pragma unroll
    for (int r = 0; r < 8; ++r) {
      int o = ot * 16 + half * 8 + r;
      gb[(size_t)o * N_DIM + n0 + col] = (__bf16)(acc[r] + bG[o]);
    }
  }
}

// ---------------- fused flash attention: y[B][N][O] = softmax(theta phi^T) g ----------------
__global__ void __launch_bounds__(128)
attn_kernel(const __bf16* __restrict__ theta, const __bf16* __restrict__ phi,
            const __bf16* __restrict__ g, __bf16* __restrict__ y) {
  int b = blockIdx.y;
  int wave = threadIdx.x >> 5, lane = threadIdx.x & 31;
  int half = lane >> 4, col = lane & 15;
  int q0 = blockIdx.x * 64 + wave * 16;
  const __bf16* Q = theta + (size_t)b * N_DIM * O_DIM;
  const __bf16* K = phi   + (size_t)b * N_DIM * O_DIM;
  const __bf16* V = g     + (size_t)b * O_DIM * N_DIM;
  const v8f vzero = {};

  v16bf qa[3];
#pragma unroll
  for (int kk = 0; kk < 3; ++kk) qa[kk] = load_a_frag(Q, O_DIM, q0, kk * 32);

  v8f acc[6];
#pragma unroll
  for (int t = 0; t < 6; ++t) acc[t] = vzero;
  float mrow[8], lrow[8];
#pragma unroll
  for (int r = 0; r < 8; ++r) { mrow[r] = -1e30f; lrow[r] = 0.f; }

  // per-wave P staging buffer (accumulator layout -> A-fragment layout)
  __shared__ __align__(16) __bf16 plds[4][16][64];
  __bf16(*pw)[64] = plds[wave];

  for (int m0 = 0; m0 < N_DIM; m0 += 64) {
    if (m0 + 64 < N_DIM) {
      __builtin_prefetch(K + (size_t)(m0 + 64) * O_DIM, 0, 1);
      __builtin_prefetch(V + m0 + 64, 0, 1);
    }
    // S = Q K^T chunk: 16 x 64
    v8f s[4];
#pragma unroll
    for (int t = 0; t < 4; ++t) {
      s[t] = vzero;
#pragma unroll
      for (int kk = 0; kk < 3; ++kk)
        s[t] = wmma_bf16(qa[kk], load_bt_frag(K, O_DIM, m0 + t * 16, kk * 32), s[t]);
    }
    // online softmax: row M lives in VGPR (M&7) across a 16-lane half
#pragma unroll
    for (int r = 0; r < 8; ++r) {
      float v = fmaxf(fmaxf(s[0][r], s[1][r]), fmaxf(s[2][r], s[3][r]));
#pragma unroll
      for (int d = 1; d < 16; d <<= 1) v = fmaxf(v, __shfl_xor(v, d, 32));
      float nm = fmaxf(mrow[r], v);
      float sc = __expf(mrow[r] - nm);
      mrow[r] = nm;
      lrow[r] *= sc;
#pragma unroll
      for (int t = 0; t < 6; ++t) acc[t][r] *= sc;
    }
    // P = exp(S - m): stash as bf16 in LDS (same-wave DS ops are in order)
#pragma unroll
    for (int t = 0; t < 4; ++t) {
#pragma unroll
      for (int r = 0; r < 8; ++r) {
        float p = __expf(s[t][r] - mrow[r]);
        s[t][r] = p;
        pw[half * 8 + r][t * 16 + col] = (__bf16)p;
      }
    }
#pragma unroll
    for (int r = 0; r < 8; ++r) {
      float v = (s[0][r] + s[1][r]) + (s[2][r] + s[3][r]);
#pragma unroll
      for (int d = 1; d < 16; d <<= 1) v += __shfl_xor(v, d, 32);
      lrow[r] += v;
    }
    // acc += P * g  (K = 64 -> two k-steps, 6 o-tiles)
#pragma unroll
    for (int kk = 0; kk < 2; ++kk) {
      v16bf pa = load_a_frag(&pw[0][0], 64, 0, kk * 32);
#pragma unroll
      for (int t = 0; t < 6; ++t)
        acc[t] = wmma_bf16(pa, load_bt_frag(V, N_DIM, t * 16, m0 + kk * 32), acc[t]);
    }
  }
  __bf16* yb = y + (size_t)b * N_DIM * O_DIM;
#pragma unroll
  for (int t = 0; t < 6; ++t) {
#pragma unroll
    for (int r = 0; r < 8; ++r) {
      float v = acc[t][r] / lrow[r];
      yb[(size_t)(q0 + half * 8 + r) * O_DIM + t * 16 + col] = (__bf16)v;
    }
  }
}

// ---------------- out = W y + b + x ----------------
__global__ void __launch_bounds__(128)
outproj_kernel(const __bf16* __restrict__ wW, const float* __restrict__ bW,
               const __bf16* __restrict__ y, const float* __restrict__ x,
               float* __restrict__ out) {
  int b = blockIdx.y;
  int wave = threadIdx.x >> 5, lane = threadIdx.x & 31;
  int half = lane >> 4, col = lane & 15;
  int n0 = blockIdx.x * 64 + wave * 16;
  const __bf16* yb = y + (size_t)b * N_DIM * O_DIM;
  const float* xb = x + (size_t)b * C_DIM * N_DIM;
  float* ob = out + (size_t)b * C_DIM * N_DIM;
  const v8f vzero = {};

  v16bf yB[3];
#pragma unroll
  for (int kk = 0; kk < 3; ++kk) yB[kk] = load_bt_frag(yb, O_DIM, n0, kk * 32);

#pragma unroll
  for (int ct = 0; ct < 12; ++ct) {
    v8f acc = vzero;
#pragma unroll
    for (int kk = 0; kk < 3; ++kk)
      acc = wmma_bf16(load_a_frag(wW, O_DIM, ct * 16, kk * 32), yB[kk], acc);
#pragma unroll
    for (int r = 0; r < 8; ++r) {
      int c = ct * 16 + half * 8 + r;
      size_t idx = (size_t)c * N_DIM + n0 + col;
      ob[idx] = acc[r] + bW[c] + xb[idx];
    }
  }
}

// ---------------- host launcher ----------------
extern "C" void kernel_launch(void* const* d_in, const int* in_sizes, int n_in,
                              void* d_out, int out_size, void* d_ws, size_t ws_size,
                              hipStream_t stream) {
  const float* x       = (const float*)d_in[0];
  const float* g_w     = (const float*)d_in[1];
  const float* g_b     = (const float*)d_in[2];
  const float* theta_w = (const float*)d_in[3];
  const float* theta_b = (const float*)d_in[4];
  const float* phi_w   = (const float*)d_in[5];
  const float* phi_b   = (const float*)d_in[6];
  const float* W_w     = (const float*)d_in[7];
  const float* W_b     = (const float*)d_in[8];
  float* out = (float*)d_out;

  // workspace carve-up (bf16 buffers), 256B aligned
  char* ws = (char*)d_ws;
  size_t off = 0;
  auto alloc = [&](size_t bytes) {
    char* p = ws + off;
    off += (bytes + 255) & ~(size_t)255;
    return p;
  };
  __bf16* xT    = (__bf16*)alloc((size_t)B_DIM * N_DIM * C_DIM * 2);  // 12.6 MB
  __bf16* thetaB= (__bf16*)alloc((size_t)B_DIM * N_DIM * O_DIM * 2);  // 6.3 MB
  __bf16* phiB  = (__bf16*)alloc((size_t)B_DIM * N_DIM * O_DIM * 2);
  __bf16* gB    = (__bf16*)alloc((size_t)B_DIM * O_DIM * N_DIM * 2);
  __bf16* yB    = (__bf16*)alloc((size_t)B_DIM * N_DIM * O_DIM * 2);
  __bf16* wTh   = (__bf16*)alloc((size_t)O_DIM * C_DIM * 2);
  __bf16* wPh   = (__bf16*)alloc((size_t)O_DIM * C_DIM * 2);
  __bf16* wG    = (__bf16*)alloc((size_t)O_DIM * C_DIM * 2);
  __bf16* wW    = (__bf16*)alloc((size_t)C_DIM * O_DIM * 2);
  (void)ws_size; (void)n_in; (void)in_sizes; (void)out_size;

  const int wn = O_DIM * C_DIM;
  cvt_f32_bf16<<<(wn + 255) / 256, 256, 0, stream>>>(theta_w, wTh, wn);
  cvt_f32_bf16<<<(wn + 255) / 256, 256, 0, stream>>>(phi_w,   wPh, wn);
  cvt_f32_bf16<<<(wn + 255) / 256, 256, 0, stream>>>(g_w,     wG,  wn);
  cvt_f32_bf16<<<(wn + 255) / 256, 256, 0, stream>>>(W_w,     wW,  wn);

  transpose_x<<<dim3(N_DIM / 32, C_DIM / 32, B_DIM), dim3(32, 8), 0, stream>>>(x, xT);

  proj_kernel<<<dim3(N_DIM / 64, B_DIM), 128, 0, stream>>>(
      xT, wTh, theta_b, wPh, phi_b, wG, g_b, thetaB, phiB, gB);

  attn_kernel<<<dim3(N_DIM / 64, B_DIM), 128, 0, stream>>>(thetaB, phiB, gB, yB);

  outproj_kernel<<<dim3(N_DIM / 64, B_DIM), 128, 0, stream>>>(wW, W_b, yB, x, out);
}
